// GeneratorTransformerXL_9809705304258
// MI455X (gfx1250) — matrix operand order repaired and checked by reading, not compiled, not run on hardware
//
#include <hip/hip_runtime.h>

// ============================================================================
// TransformerXL forward on gfx1250 (MI455X), bf16 WMMA GEMM path.
// Compute-bound (~1.4 TFLOP vs ~0.65 GB weight traffic) -> all GEMMs through
// v_wmma_f32_16x16x32_bf16. Interior tiles use double-buffered LDS filled by
// GLOBAL_LOAD_ASYNC_TO_LDS_B128 (ASYNCcnt) to overlap HBM fetch with WMMA.
// ============================================================================

typedef __bf16 bf16;
typedef float  v8f   __attribute__((ext_vector_type(8)));
typedef bf16   v16bf __attribute__((ext_vector_type(16)));

#define LQ    513      // qlen = S+1
#define LB    8        // batch
#define LKL   523      // klen = qlen + MEM
#define LKLP  544      // klen padded to mult of 32 (GEMM K)
#define LD    1024     // d_model
#define LH    8        // heads
#define LDH   128      // d_head
#define LDI   4096     // d_inner
#define LMEM  10
#define LLAYERS 12
#define LLAB  1001
#define LS    512
#define LQB   (LQ*LB)  // 4104
#define LKB   (LKL*LB) // 4184

#define LDSROW 40                  // bf16 per LDS tile row (32 + 8 pad, 80B: bank-conflict-free)
#define LDSBUF (128 * LDSROW)      // elems per tile buffer
#define LDSBUFB (LDSBUF * 2)       // bytes per tile buffer (10240)

union FragU { v16bf v; uint4 u[2]; };

// generic -> LDS byte offset (addrspacecast + ptrtoint; AS3 pointers are 32-bit)
typedef __attribute__((address_space(3))) const void lds_cvoid;
__device__ __forceinline__ unsigned lds_addr(const void* p)
{
    return (unsigned)(unsigned long long)(lds_cvoid*)p;
}

// one async 16B global->LDS copy, tracked by ASYNCcnt
__device__ __forceinline__ void async_b128(unsigned ldsoff, const bf16* g)
{
    unsigned long long ga = (unsigned long long)g;
    asm volatile("global_load_async_to_lds_b128 %0, %1, off" :: "v"(ldsoff), "v"(ga) : "memory");
}

// fragment loads + 8 WMMAs for one 32-K step
__device__ __forceinline__ void mma_tile(const bf16* la, const bf16* lb, v8f acc[2][4],
                                         int wr, int wc, int lrow, int khalf)
{
    // A fragment (16x32 bf16): e[i<8] = K=h*8+i ; e[i>=8] = K=16+h*8+(i-8)
    v16bf af[2];
    #pragma unroll
    for (int mt = 0; mt < 2; ++mt) {
        int row = wr * 32 + mt * 16 + lrow;
        FragU f;
        f.u[0] = *(const uint4*)(la + row * LDSROW + khalf * 8);
        f.u[1] = *(const uint4*)(la + row * LDSROW + 16 + khalf * 8);
        af[mt] = f.v;
    }
    // B fragment (32x16): e[i<8] = K=h*8+i ; e[i>=8] = K=16+h*8+(i-8), col = lane&15
    #pragma unroll
    for (int nt = 0; nt < 4; ++nt) {
        int row = wc * 64 + nt * 16 + lrow;   // row of W == output column
        FragU f;
        f.u[0] = *(const uint4*)(lb + row * LDSROW + khalf * 8);
        f.u[1] = *(const uint4*)(lb + row * LDSROW + 16 + khalf * 8);
        v16bf bv = f.v;
        #pragma unroll
        for (int mt = 0; mt < 2; ++mt)
            acc[mt][nt] = __builtin_amdgcn_wmma_f32_16x16x32_bf16(
                false, af[mt], false, bv, (short)0, acc[mt][nt], false, false);
    }
}

// ---------------------------------------------------------------------------
// Batched WMMA GEMM: C[z] = A[z] (MxK) * W[wz] (NxK)^T  (+bias, relu)
// A,W bf16 row-major with K contiguous. Outputs f32 and/or bf16.
// wz = (wmod>0) ? z%wmod : z   (lets BD GEMM index per-head r_k weights)
// ---------------------------------------------------------------------------
__global__ __launch_bounds__(256)
void k_wmma_gemm(const bf16* __restrict__ A, long long sAz, int lda,
                 const bf16* __restrict__ W, long long sWz, int wmod, int ldw,
                 float* __restrict__ Cf, long long sCfz, int ldcf,
                 bf16* __restrict__ Cb, long long sCbz, int ldcb,
                 const float* __restrict__ bias,
                 int M, int N, int K, int relu)
{
    __shared__ __align__(16) bf16 lA[2][LDSBUF];   // double-buffered 128x32 tiles
    __shared__ __align__(16) bf16 lB[2][LDSBUF];

    const int z = blockIdx.z;
    const bf16* Az = A + (long long)z * sAz;
    const bf16* Wz = W + (long long)(wmod > 0 ? (z % wmod) : z) * sWz;
    const int tid  = threadIdx.x;
    const int lane = tid & 31, wave = tid >> 5;
    const int wr = wave >> 1, wc = wave & 1;          // wave grid 4x2 -> 32x64 per wave
    const int m0 = blockIdx.y * 128, n0 = blockIdx.x * 128;
    const int lrow = lane & 15, khalf = lane >> 4;

    v8f acc[2][4] = {};

    const bool full = (m0 + 128 <= M) && (n0 + 128 <= N);
    if (full) {
        // ---------------- interior tile: async double-buffered pipeline ----
        const int c0 = tid, c1 = tid + 256;           // this thread's 2 chunks per tile
        const int r0 = c0 >> 2, k80 = (c0 & 3) * 8;
        const int r1 = c1 >> 2, k81 = (c1 & 3) * 8;
        const bf16* ga0 = Az + (long long)(m0 + r0) * lda + k80;
        const bf16* ga1 = Az + (long long)(m0 + r1) * lda + k81;
        const bf16* gb0 = Wz + (long long)(n0 + r0) * ldw + k80;
        const bf16* gb1 = Wz + (long long)(n0 + r1) * ldw + k81;
        const unsigned la0 = lds_addr(&lA[0][r0 * LDSROW + k80]);
        const unsigned la1 = lds_addr(&lA[0][r1 * LDSROW + k81]);
        const unsigned lb0 = lds_addr(&lB[0][r0 * LDSROW + k80]);
        const unsigned lb1 = lds_addr(&lB[0][r1 * LDSROW + k81]);

        auto issue = [&](int buf, int k0) {
            unsigned d = buf ? (unsigned)LDSBUFB : 0u;
            async_b128(la0 + d, ga0 + k0);
            async_b128(la1 + d, ga1 + k0);
            async_b128(lb0 + d, gb0 + k0);
            async_b128(lb1 + d, gb1 + k0);
        };

        const int nIter = K >> 5;
        issue(0, 0);
        for (int i = 0; i < nIter; ++i) {
            if (i + 1 < nIter) {
                issue((i + 1) & 1, (i + 1) << 5);     // prefetch next tile
                asm volatile("s_wait_asynccnt 0x4" ::: "memory");  // tile i landed
            } else {
                asm volatile("s_wait_asynccnt 0x0" ::: "memory");
            }
            __syncthreads();
            mma_tile(lA[i & 1], lB[i & 1], acc, wr, wc, lrow, khalf);
            __syncthreads();                          // all waves done before rewrite
        }
    } else {
        // ---------------- edge tile: guarded synchronous path --------------
        for (int k0 = 0; k0 < K; k0 += 32) {
            #pragma unroll
            for (int pass = 0; pass < 2; ++pass) {
                int c  = tid + pass * 256;            // 0..511 chunks
                int r  = c >> 2;
                int k8 = (c & 3) * 8;
                uint4 va = make_uint4(0u, 0u, 0u, 0u);
                uint4 vb = make_uint4(0u, 0u, 0u, 0u);
                if (m0 + r < M) va = *(const uint4*)(Az + (long long)(m0 + r) * lda + k0 + k8);
                if (n0 + r < N) vb = *(const uint4*)(Wz + (long long)(n0 + r) * ldw + k0 + k8);
                *(uint4*)(&lA[0][r * LDSROW + k8]) = va;
                *(uint4*)(&lB[0][r * LDSROW + k8]) = vb;
            }
            __syncthreads();
            mma_tile(lA[0], lB[0], acc, wr, wc, lrow, khalf);
            __syncthreads();
        }
    }

    // epilogue: C layout per ISA: e[r] -> M = r + 8*(lane>>4), N = lane&15
    #pragma unroll
    for (int mt = 0; mt < 2; ++mt) {
        #pragma unroll
        for (int nt = 0; nt < 4; ++nt) {
            int col = n0 + wc * 64 + nt * 16 + lrow;
            if (col >= N) continue;
            float bv = bias ? bias[col] : 0.0f;
            #pragma unroll
            for (int r = 0; r < 8; ++r) {
                int row = m0 + wr * 32 + mt * 16 + khalf * 8 + r;
                if (row >= M) continue;
                float v = acc[mt][nt][r] + bv;
                if (relu) v = fmaxf(v, 0.0f);
                if (Cf) Cf[(long long)z * sCfz + (long long)row * ldcf + col] = v;
                if (Cb) Cb[(long long)z * sCbz + (long long)row * ldcb + col] = (bf16)v;
            }
        }
    }
}

// ---------------------------------------------------------------------------
// elementwise helpers
// ---------------------------------------------------------------------------
__global__ void k_f32_to_bf16(const float* __restrict__ in, bf16* __restrict__ out, int n)
{
    int i = blockIdx.x * blockDim.x + threadIdx.x;
    if (i < n) out[i] = (bf16)in[i];
}

__global__ void k_zero_bf16(bf16* __restrict__ p, int n)
{
    int i = blockIdx.x * blockDim.x + threadIdx.x;
    if (i < n) p[i] = (bf16)0.0f;
}

// embedding gather * sqrt(D); writes residual stream (f32) and QKV input (bf16)
__global__ void k_embed(const int* __restrict__ inputs, const int* __restrict__ relation,
                        const float* __restrict__ emb, float* __restrict__ h,
                        bf16* __restrict__ catb)
{
    int t = blockIdx.x;                 // t = i*B + b
    int i = t >> 3, b = t & 7;
    int id = (i == 0) ? relation[b] : inputs[b * LS + (i - 1)];
    const float* e = emb + (long long)id * LD;
    for (int d = threadIdx.x; d < LD; d += blockDim.x) {
        float v = e[d] * 32.0f;         // sqrt(1024)
        h[(long long)t * LD + d] = v;
        catb[((long long)(LMEM * LB + t)) * LD + d] = (bf16)v;
    }
}

// sinusoidal relative position embedding (bf16, GEMM A-operand)
__global__ void k_posemb(bf16* __restrict__ posb)
{
    int p = blockIdx.x;                 // 0..KL-1
    float pos = fminf((float)(LKL - 1 - p), 1000.0f);
    for (int d2 = threadIdx.x; d2 < LD / 2; d2 += blockDim.x) {
        float invf = __expf(-((float)(2 * d2) / (float)LD) * 9.210340371976184f); // ln(10000)
        float a = pos * invf;
        posb[(long long)p * LD + d2]          = (bf16)sinf(a);
        posb[(long long)p * LD + LD / 2 + d2] = (bf16)cosf(a);
    }
}

// split qkv rows -> per-(b,h) operands: qA=q+u, qB=q+v, k, v^T (padded cols zeroed once)
__global__ void k_qkv_pack(const bf16* __restrict__ qkv,
                           const float* __restrict__ rwb, const float* __restrict__ rrb,
                           bf16* __restrict__ qA, bf16* __restrict__ qB,
                           bf16* __restrict__ kb, bf16* __restrict__ vt)
{
    int t = blockIdx.x;                 // qkv row = j*B + b
    int j = t >> 3, b = t & 7;
    const bf16* row = qkv + (long long)t * (3 * LD);
    for (int c = threadIdx.x; c < LD; c += blockDim.x) {
        int hh = c >> 7, d = c & 127;
        int z  = b * LH + hh;
        kb[((long long)z * LKL + j) * LDH + d]  = row[LD + c];
        vt[((long long)z * LDH + d) * LKLP + j] = row[2 * LD + c];
        if (j >= LMEM) {
            int i = j - LMEM;
            float qv = (float)row[c];
            qA[((long long)z * LQ + i) * LDH + d] = (bf16)(qv + rwb[c]);
            qB[((long long)z * LQ + i) * LDH + d] = (bf16)(qv + rrb[c]);
        }
    }
}

// score = (AC + rel_shift(BD)) * scale; causal+mem mask; softmax; write prob bf16
// rel_shift folded analytically: BD[i,j] = BDpre[i, j + Q-1-i]; valid keys: j <= i+MEM
__global__ __launch_bounds__(256)
void k_softmax(const float* __restrict__ AC, const float* __restrict__ BD,
               bf16* __restrict__ P)
{
    int i = blockIdx.x;                 // query row
    int z = blockIdx.y;                 // b*H + h
    const float* ac = AC + ((long long)z * LQ + i) * LKL;
    const float* bd = BD + ((long long)z * LQ + i) * LKL;
    bf16* pr = P + ((long long)z * LQ + i) * LKLP;
    const int nv = i + LMEM + 1;        // valid key count (<=523)
    const int shift = LQ - 1 - i;
    const float scale = 0.088388347648318447f;   // 1/sqrt(128)

    float sv[3];
    float mx = -3.4e38f;
    int cnt = 0;
    for (int j = threadIdx.x; j < nv; j += 256) {
        float s = (ac[j] + bd[j + shift]) * scale;
        sv[cnt++] = s;
        mx = fmaxf(mx, s);
    }
    __shared__ float red[256];
    red[threadIdx.x] = mx; __syncthreads();
    for (int o = 128; o > 0; o >>= 1) {
        if (threadIdx.x < o) red[threadIdx.x] = fmaxf(red[threadIdx.x], red[threadIdx.x + o]);
        __syncthreads();
    }
    mx = red[0]; __syncthreads();

    float lsum = 0.0f; cnt = 0;
    for (int j = threadIdx.x; j < nv; j += 256) {
        sv[cnt] = __expf(sv[cnt] - mx);
        lsum += sv[cnt]; ++cnt;
    }
    red[threadIdx.x] = lsum; __syncthreads();
    for (int o = 128; o > 0; o >>= 1) {
        if (threadIdx.x < o) red[threadIdx.x] += red[threadIdx.x + o];
        __syncthreads();
    }
    float inv = 1.0f / red[0];

    cnt = 0;
    for (int j = threadIdx.x; j < LKLP; j += 256) {
        float v = 0.0f;
        if (j < nv) v = sv[cnt++] * inv;
        pr[j] = (bf16)v;
    }
}

// h = LayerNorm(h + delta); also emit bf16 copy for the next GEMM
__global__ __launch_bounds__(256)
void k_add_ln(float* __restrict__ h, const float* __restrict__ delta,
              const float* __restrict__ g, const float* __restrict__ bt,
              bf16* __restrict__ outb)
{
    long long row = blockIdx.x;
    float* hr = h + row * LD;
    const float* dr = delta + row * LD;
    float x[4];
    float s = 0.0f;
    #pragma unroll
    for (int k = 0; k < 4; ++k) {
        int d = threadIdx.x + k * 256;
        x[k] = hr[d] + dr[d];
        s += x[k];
    }
    __shared__ float red[256];
    red[threadIdx.x] = s; __syncthreads();
    for (int o = 128; o > 0; o >>= 1) { if (threadIdx.x < o) red[threadIdx.x] += red[threadIdx.x + o]; __syncthreads(); }
    float mu = red[0] * (1.0f / LD); __syncthreads();
    float v = 0.0f;
    #pragma unroll
    for (int k = 0; k < 4; ++k) { float dd = x[k] - mu; v += dd * dd; }
    red[threadIdx.x] = v; __syncthreads();
    for (int o = 128; o > 0; o >>= 1) { if (threadIdx.x < o) red[threadIdx.x] += red[threadIdx.x + o]; __syncthreads(); }
    float rstd = rsqrtf(red[0] * (1.0f / LD) + 1e-5f);
    #pragma unroll
    for (int k = 0; k < 4; ++k) {
        int d = threadIdx.x + k * 256;
        float y = (x[k] - mu) * rstd * g[d] + bt[d];
        hr[d] = y;
        outb[row * LD + d] = (bf16)y;
    }
}

// head logits [q*B+b, LAB] (q<512) -> d_out [b, q, LAB]
__global__ void k_head_copy(const float* __restrict__ c, float* __restrict__ out)
{
    int t = blockIdx.x;                 // t = q*B + b, q < 512
    int q = t >> 3, b = t & 7;
    for (int l = threadIdx.x; l < LLAB; l += blockDim.x)
        out[((long long)b * LS + q) * LLAB + l] = c[(long long)t * LLAB + l];
}

// ---------------------------------------------------------------------------
static inline void gemm(hipStream_t s,
                        const bf16* A, long long sAz, int lda,
                        const bf16* W, long long sWz, int wmod, int ldw,
                        float* Cf, long long sCfz, int ldcf,
                        bf16* Cb, long long sCbz, int ldcb,
                        const float* bias, int M, int N, int K, int relu, int batch)
{
    dim3 g((N + 127) / 128, (M + 127) / 128, batch);
    k_wmma_gemm<<<g, 256, 0, s>>>(A, sAz, lda, W, sWz, wmod, ldw,
                                  Cf, sCfz, ldcf, Cb, sCbz, ldcb,
                                  bias, M, N, K, relu);
}

extern "C" void kernel_launch(void* const* d_in, const int* in_sizes, int n_in,
                              void* d_out, int out_size, void* d_ws, size_t ws_size,
                              hipStream_t stream)
{
    (void)in_sizes; (void)n_in; (void)out_size; (void)ws_size;

    const int*   inp   = (const int*)d_in[0];
    const int*   rel   = (const int*)d_in[1];
    const float* emb   = (const float*)d_in[2];
    const float* qkv_w = (const float*)d_in[3];
    const float* o_w   = (const float*)d_in[4];
    const float* r_w   = (const float*)d_in[5];
    const float* rwb   = (const float*)d_in[6];
    const float* rrb   = (const float*)d_in[7];
    const float* ln1g  = (const float*)d_in[8];
    const float* ln1b  = (const float*)d_in[9];
    const float* f1w   = (const float*)d_in[10];
    const float* f1b   = (const float*)d_in[11];
    const float* f2w   = (const float*)d_in[12];
    const float* f2b   = (const float*)d_in[13];
    const float* ln2g  = (const float*)d_in[14];
    const float* ln2b  = (const float*)d_in[15];
    const float* hw    = (const float*)d_in[16];
    const float* hbias = (const float*)d_in[17];
    float* out = (float*)d_out;

    // ---- workspace carve (~357 MB) ----
    char* base = (char*)d_ws;
    size_t off = 0;
    auto alloc = [&](size_t bytes) -> void* {
        void* p = base + off;
        off += (bytes + 255) & ~(size_t)255;
        return p;
    };
    float* h      = (float*)alloc(4ull * LQB * LD);          // residual stream
    bf16*  catb   = (bf16*) alloc(2ull * LKB * LD);          // [mems(0); h] bf16 (QKV input)
    bf16*  qkvb   = (bf16*) alloc(2ull * LKB * 3 * LD);
    bf16*  posb   = (bf16*) alloc(2ull * LKL * LD);
    bf16*  rkb    = (bf16*) alloc(2ull * LKL * LD);          // r_k bf16 [KL, H*DH]
    bf16*  qAb    = (bf16*) alloc(2ull * 64 * LQ * LDH);     // q + r_w_bias
    bf16*  qBb    = (bf16*) alloc(2ull * 64 * LQ * LDH);     // q + r_r_bias
    bf16*  kbf    = (bf16*) alloc(2ull * 64 * LKL * LDH);
    bf16*  vtb    = (bf16*) alloc(2ull * 64 * LDH * LKLP);   // v transposed, K-padded
    float* scoreA = (float*)alloc(4ull * 64 * LQ * LKL);     // AC
    float* scoreB = (float*)alloc(4ull * 64 * LQ * LKL);     // BD (pre-shift)
    bf16*  Pb     = (bf16*) alloc(2ull * 64 * LQ * LKLP);    // probs
    bf16*  attnb  = (bf16*) alloc(2ull * LQB * LD);
    float* oout   = (float*)alloc(4ull * LQB * LD);
    bf16*  hbf    = (bf16*) alloc(2ull * LQB * LD);
    bf16*  ffact  = (bf16*) alloc(2ull * LQB * LDI);
    bf16*  wqkv   = (bf16*) alloc(2ull * 3 * LD * LD);
    bf16*  wo     = (bf16*) alloc(2ull * LD * LD);
    bf16*  wr     = (bf16*) alloc(2ull * LD * LD);
    bf16*  wf1    = (bf16*) alloc(2ull * LDI * LD);
    bf16*  wf2    = (bf16*) alloc(2ull * LD * LDI);
    bf16*  whead  = (bf16*) alloc(2ull * LLAB * LD);
    float* headC  = scoreA;   // reuse (attention done by then)
    float* ff2out = oout;     // reuse (consumed by LN1 before FF2)

    const int T = 256;
    // init: zero mem rows of cat, zero v^T (covers K-pad region every call)
    k_zero_bf16<<<(LMEM * LB * LD + T - 1) / T, T, 0, stream>>>(catb, LMEM * LB * LD);
    k_zero_bf16<<<(64 * LDH * LKLP + T - 1) / T, T, 0, stream>>>(vtb, 64 * LDH * LKLP);
    k_embed<<<LQB, T, 0, stream>>>(inp, rel, emb, h, catb);
    k_posemb<<<LKL, T, 0, stream>>>(posb);
    k_f32_to_bf16<<<(LLAB * LD + T - 1) / T, T, 0, stream>>>(hw, whead, LLAB * LD);

    for (int l = 0; l < LLAYERS; ++l) {
        // per-layer weight conversion f32 -> bf16 (tiny vs GEMM cost, saves 300MB ws)
        k_f32_to_bf16<<<(3 * LD * LD + T - 1) / T, T, 0, stream>>>(qkv_w + (size_t)l * 3 * LD * LD, wqkv, 3 * LD * LD);
        k_f32_to_bf16<<<(LD * LD + T - 1) / T, T, 0, stream>>>(o_w + (size_t)l * LD * LD, wo, LD * LD);
        k_f32_to_bf16<<<(LD * LD + T - 1) / T, T, 0, stream>>>(r_w + (size_t)l * LD * LD, wr, LD * LD);
        k_f32_to_bf16<<<(LDI * LD + T - 1) / T, T, 0, stream>>>(f1w + (size_t)l * LDI * LD, wf1, LDI * LD);
        k_f32_to_bf16<<<(LD * LDI + T - 1) / T, T, 0, stream>>>(f2w + (size_t)l * LD * LDI, wf2, LD * LDI);

        // QKV: [KB, D] x [3D, D]^T -> bf16 [KB, 3D]
        gemm(stream, catb, 0, LD, wqkv, 0, 0, LD,
             nullptr, 0, 0, qkvb, 0, 3 * LD, nullptr, LKB, 3 * LD, LD, 0, 1);
        // r_k: [KL, D] x [D, D]^T -> bf16 [KL, D]
        gemm(stream, posb, 0, LD, wr, 0, 0, LD,
             nullptr, 0, 0, rkb, 0, LD, nullptr, LKL, LD, LD, 0, 1);
        k_qkv_pack<<<LKB, T, 0, stream>>>(qkvb, rwb + (size_t)l * LD, rrb + (size_t)l * LD,
                                          qAb, qBb, kbf, vtb);
        // AC[z] = (q+u) k^T : batched 64x [Q,DH]x[KL,DH]^T -> f32 [Q,KL]
        gemm(stream, qAb, (long long)LQ * LDH, LDH, kbf, (long long)LKL * LDH, 0, LDH,
             scoreA, (long long)LQ * LKL, LKL, nullptr, 0, 0, nullptr, LQ, LKL, LDH, 0, 64);
        // BD[z] = (q+v) r_k[h]^T : per-head weights via wmod=H (wz = z%8 = h)
        gemm(stream, qBb, (long long)LQ * LDH, LDH, rkb, LDH, LH, LD,
             scoreB, (long long)LQ * LKL, LKL, nullptr, 0, 0, nullptr, LQ, LKL, LDH, 0, 64);
        dim3 sg(LQ, 64);
        k_softmax<<<sg, T, 0, stream>>>(scoreA, scoreB, Pb);
        // attn = P V : [Q,KLP]x[DH,KLP]^T, scattered into [Q,B,D] bf16 (col base = 128*z)
        gemm(stream, Pb, (long long)LQ * LKLP, LKLP, vtb, (long long)LDH * LKLP, 0, LKLP,
             nullptr, 0, 0, attnb, 128, LB * LD, nullptr, LQ, LDH, LKLP, 0, 64);
        // O projection
        gemm(stream, attnb, 0, LD, wo, 0, 0, LD,
             oout, 0, LD, nullptr, 0, 0, nullptr, LQB, LD, LD, 0, 1);
        k_add_ln<<<LQB, T, 0, stream>>>(h, oout, ln1g + (size_t)l * LD, ln1b + (size_t)l * LD, hbf);
        // FF1 (+bias, relu) -> bf16
        gemm(stream, hbf, 0, LD, wf1, 0, 0, LD,
             nullptr, 0, 0, ffact, 0, LDI, f1b + (size_t)l * LDI, LQB, LDI, LD, 1, 1);
        // FF2 (+bias) -> f32
        gemm(stream, ffact, 0, LDI, wf2, 0, 0, LDI,
             ff2out, 0, LD, nullptr, 0, 0, f2b + (size_t)l * LD, LQB, LD, LDI, 0, 1);
        // LN2 writes residual f32 and next layer's bf16 QKV input (cat rows MEM*B..)
        k_add_ln<<<LQB, T, 0, stream>>>(h, ff2out, ln2g + (size_t)l * LD, ln2b + (size_t)l * LD,
                                        catb + (size_t)LMEM * LB * LD);
    }

    // head: only q<512 needed -> M = 4096 rows (t = q*B+b)
    gemm(stream, catb + (size_t)LMEM * LB * LD, 0, LD, whead, 0, 0, LD,
         headC, 0, LLAB, nullptr, 0, 0, hbias, LS * LB, LLAB, LD, 0, 1);
    k_head_copy<<<LS * LB, T, 0, stream>>>(headC, out);
}
